// CCPM_52853867544632
// MI455X (gfx1250) — matrix operand order repaired and checked by reading, not compiled, not run on hardware
//
#include <hip/hip_runtime.h>

typedef __attribute__((ext_vector_type(16))) _Float16 v16h;
typedef __attribute__((ext_vector_type(8)))  _Float16 v8h;
typedef __attribute__((ext_vector_type(8)))  float    v8f;

#define CC_F 23
#define CC_VOCAB 100000
#define CC_D 64
#define SPB 8   // samples per block
#define BLK 512 // SPB * 64

// One conv (KW taps, IN_CH->4 channels, pad KW-1 both sides, VALID) followed
// by order-preserving top-k (lax.top_k tie semantics: kept set prefers lower
// index) and ReLU. Tap/channel loops unrolled (small); position/rank loops
// rolled to keep compile cost bounded.
template <int N_IN, int IN_CH, int KW, int K_KEEP>
__device__ __forceinline__ void conv_pool(const float* w, const float* bias,
                                          const float in[][28],
                                          float out[][28]) {
  const int n_conv = N_IN + KW - 1;
  for (int c = 0; c < 4; ++c) {
    float t[28];
    for (int p = 0; p < n_conv; ++p) {
      float acc = bias[c];
#pragma unroll
      for (int kk = 0; kk < KW; ++kk) {
        const int j = p + kk - (KW - 1);
        if (j >= 0 && j < N_IN) {
#pragma unroll
          for (int ci = 0; ci < IN_CH; ++ci)
            acc += w[(kk * IN_CH + ci) * 4 + c] * in[ci][j];
        }
      }
      t[p] = acc;
    }
    // rank of t[i] in descending order (ties -> lower index ranks higher)
    int o = 0;
    for (int i = 0; i < n_conv; ++i) {
      int r = 0;
      for (int j = 0; j < n_conv; ++j)
        r += (t[j] > t[i] || (t[j] == t[i] && j < i)) ? 1 : 0;
      if (r < K_KEEP) {
        out[c][o] = fmaxf(t[i], 0.0f);
        ++o;
      }
    }
  }
}

__global__ void ccpm_fused_kernel(
    const int* __restrict__ ids, const float* __restrict__ emb,
    const float* __restrict__ w1, const float* __restrict__ b1,
    const float* __restrict__ w2, const float* __restrict__ b2,
    const float* __restrict__ w3, const float* __restrict__ b3,
    const float* __restrict__ wd, const float* __restrict__ bd,
    float* __restrict__ out, int B) {
  __shared__ float sW1[24];   // (6,1,4)  -> [t*4+c]
  __shared__ float sW2[80];   // (5,4,4)  -> [(t*4+ci)*4+c]
  __shared__ float sW3[48];   // (3,4,4)  -> [(t*4+ci)*4+c]
  __shared__ float sB[12];    // b1|b2|b3
  __shared__ __align__(32) _Float16 sWdH[768];  // dense weights (f16)
  __shared__ int sIds[SPB][CC_F];               // per-sample ids
  // Full 16-row A panel; rows SPB..15 zeroed so the WMMA feed needs no
  // per-lane guards and can use aligned 128-bit LDS loads.
  __shared__ __align__(32) _Float16 sFeatH[16][768];

  const int tid = threadIdx.x;
  const int sl = tid >> 6;  // local sample 0..SPB-1
  const int d = tid & 63;   // row 0..63
  const int b = blockIdx.x * SPB + sl;

  // ---- cooperative staging of weights / ids into LDS ----
  for (int i = tid; i < 24; i += BLK) sW1[i] = w1[i];
  for (int i = tid; i < 80; i += BLK) sW2[i] = w2[i];
  for (int i = tid; i < 48; i += BLK) sW3[i] = w3[i];
  if (tid < 4) {
    sB[tid] = b1[tid];
    sB[4 + tid] = b2[tid];
    sB[8 + tid] = b3[tid];
  }
  for (int i = tid; i < 768; i += BLK) sWdH[i] = (_Float16)wd[i];
  // zero unused A-panel rows once
  {
    _Float16* z = &sFeatH[SPB][0];
    for (int i = tid; i < (16 - SPB) * 768; i += BLK) z[i] = (_Float16)0.0f;
  }
  if (d < CC_F && b < B) sIds[sl][d] = ids[(size_t)b * CC_F + d];
  __syncthreads();

  if (b < B) {
    float bufA[4][28];
    float bufB[4][28];

    // gather one embedding row per feature (coalesced across d)
    for (int f = 0; f < CC_F; ++f) {
      size_t off = ((size_t)f * CC_VOCAB + (size_t)sIds[sl][f]) * CC_D + d;
      bufA[0][f] = emb[off];
    }

    conv_pool<23, 1, 6, 20>(sW1, sB + 0, bufA, bufB);  // -> 20 x 4
    conv_pool<20, 4, 5, 7>(sW2, sB + 4, bufB, bufA);   // -> 7 x 4
    conv_pool<7, 4, 3, 3>(sW3, sB + 8, bufA, bufB);    // -> 3 x 4

    for (int j = 0; j < 3; ++j)
      for (int c = 0; c < 4; ++c)
        sFeatH[sl][d * 12 + j * 4 + c] = (_Float16)bufB[c][j];
  }
  __syncthreads();

  // ---- dense(768->1) + sigmoid via V_WMMA_F32_16X16X32_F16 on wave 0 ----
  // A (16x32 f16): rows = samples (rows SPB..15 zero). ISA layout: lane half
  // `hi` holds two contiguous 8-half K chunks at offsets {0,16} (lo) or
// {8,24} (hi). B (32x16 f16): wd[k] broadcast across all 16 columns (lane
  // half holds the same K chunks as A: {0,16} lo or {8,24} hi), so every column of D
  // is the GEMV result. K=768 -> 24 WMMA steps, f32 accumulate.
  if (tid < 32) {
    const int lane = tid;
    const bool hi = lane >= 16;
    const int m = hi ? lane - 16 : lane;  // A-matrix row for this lane
    const int aOff0 = hi ? 8 : 0;
    const int aOff1 = hi ? 24 : 16;
    const int bOff = hi ? 16 : 0;
    v8f acc = {};
#pragma unroll 4
    for (int k0 = 0; k0 < 768; k0 += 32) {
      union { v16h v; v8h p[2]; } ua;
      ua.p[0] = *(const v8h*)&sFeatH[m][k0 + aOff0];
      ua.p[1] = *(const v8h*)&sFeatH[m][k0 + aOff1];
      union { v16h v; v8h p[2]; } ub;
      ub.p[0] = *(const v8h*)&sWdH[k0 + aOff0];
      ub.p[1] = *(const v8h*)&sWdH[k0 + aOff1];
      const v16h bv = ub.v;
      acc = __builtin_amdgcn_wmma_f32_16x16x32_f16(false, ua.v, false, bv,
                                                   (short)0, acc, false,
                                                   false);
    }
    // D layout: VGPR r on lane 0 holds D[r][0] (samples 0..7).
    if (lane == 0) {
      const float bias = bd[0];
      const int base = blockIdx.x * SPB;
      for (int r = 0; r < SPB; ++r) {
        if (base + r < B) {
          const float v = acc[r] + bias;
          out[base + r] = 1.0f / (1.0f + __expf(-v));
        }
      }
    }
  }
}

extern "C" void kernel_launch(void* const* d_in, const int* in_sizes, int n_in,
                              void* d_out, int out_size, void* d_ws,
                              size_t ws_size, hipStream_t stream) {
  const int* ids = (const int*)d_in[0];
  const float* emb = (const float*)d_in[1];
  const float* w1 = (const float*)d_in[2];
  const float* b1 = (const float*)d_in[3];
  const float* w2 = (const float*)d_in[4];
  const float* b2 = (const float*)d_in[5];
  const float* w3 = (const float*)d_in[6];
  const float* b3 = (const float*)d_in[7];
  const float* wd = (const float*)d_in[8];
  const float* bd = (const float*)d_in[9];
  float* out = (float*)d_out;

  const int B = in_sizes[0] / CC_F;
  dim3 grid((B + SPB - 1) / SPB);
  dim3 block(BLK);
  ccpm_fused_kernel<<<grid, block, 0, stream>>>(ids, emb, w1, b1, w2, b2, w3,
                                                b3, wd, bd, out, B);
}
